// SpatialCrossAttention_58171037057722
// MI455X (gfx1250) — compile-verified, matrix operand order checked by reading, not run on hardware
//
#include <hip/hip_runtime.h>
#include <hip/hip_fp16.h>

#define N_CAM 6
#define L_Q   40000
#define CH    256
#define HF    116
#define WF    200
#define HW    (HF * WF)

#define QPB        16    // queries per block (16 -> WMMA M tile)
#define LDS_STRIDE 260   // 260 % 64 banks = 4 -> conflict-free column reads

typedef __attribute__((ext_vector_type(2)))  float    v2f;
typedef __attribute__((ext_vector_type(8)))  float    v8f;
typedef __attribute__((ext_vector_type(16))) _Float16 v16h;

__launch_bounds__(256)
__global__ void sca_fused_kernel(const float* __restrict__ query,
                                 const float* __restrict__ value,
                                 const float* __restrict__ refpts,
                                 const float* __restrict__ bev_mask,
                                 const float* __restrict__ W_out,
                                 const float* __restrict__ b_out,
                                 float* __restrict__ out)
{
    __shared__ float ctx[QPB][LDS_STRIDE];

    const int lane = threadIdx.x & 31;
    const int wave = threadIdx.x >> 5;
    const int l0   = blockIdx.x * QPB;

    // ---------------- Phase 1: fused gather + camera attention ----------------
    // Each wave owns 2 queries; lane handles channels c = i*32 + lane (i=0..7).
    for (int t = 0; t < 2; ++t) {
        const int qlocal = wave * 2 + t;
        const int q      = l0 + qlocal;

        float qv[8];
        #pragma unroll
        for (int i = 0; i < 8; ++i)
            qv[i] = query[q * CH + i * 32 + lane];   // coalesced 128B

        float feats[N_CAM][8];
        float score[N_CAM];

        #pragma unroll
        for (int n = 0; n < N_CAM; ++n) {
            const float mv = bev_mask[n * L_Q + q];  // wave-uniform
            if (mv == 0.0f) {                        // masked cam: feats=0, score=0
                score[n] = 0.0f;
                #pragma unroll
                for (int i = 0; i < 8; ++i) feats[n][i] = 0.0f;
                continue;
            }
            const float2 g = ((const float2*)refpts)[n * L_Q + q];
            // grid_sample, align_corners=False
            const float x = (g.x + 1.0f) * (0.5f * WF) - 0.5f;
            const float y = (g.y + 1.0f) * (0.5f * HF) - 0.5f;
            const float x0f = floorf(x), y0f = floorf(y);
            const int   x0 = (int)x0f,  y0 = (int)y0f;
            const int   x1 = x0 + 1,    y1 = y0 + 1;
            const float wx1 = x - x0f, wx0 = 1.0f - wx1;
            const float wy1 = y - y0f, wy0 = 1.0f - wy1;
            const bool vx0 = (x0 >= 0) & (x0 < WF);
            const bool vx1 = (x1 >= 0) & (x1 < WF);
            const bool vy0 = (y0 >= 0) & (y0 < HF);
            const bool vy1 = (y1 >= 0) & (y1 < HF);
            // fold zero-padding validity AND bev_mask into the 4 corner weights
            const float w00 = (vy0 & vx0) ? wy0 * wx0 * mv : 0.0f;
            const float w01 = (vy0 & vx1) ? wy0 * wx1 * mv : 0.0f;
            const float w10 = (vy1 & vx0) ? wy1 * wx0 * mv : 0.0f;
            const float w11 = (vy1 & vx1) ? wy1 * wx1 * mv : 0.0f;
            const int cx0 = min(max(x0, 0), WF - 1);
            const int cx1 = min(max(x1, 0), WF - 1);
            const int cy0 = min(max(y0, 0), HF - 1);
            const int cy1 = min(max(y1, 0), HF - 1);
            const int o00 = cy0 * WF + cx0;
            const int o01 = cy0 * WF + cx1;
            const int o10 = cy1 * WF + cx0;
            const int o11 = cy1 * WF + cx1;

            const float* vbase = value + (n * CH) * HW;
            float dot = 0.0f;
            #pragma unroll
            for (int i = 0; i < 8; ++i) {
                const int    c  = i * 32 + lane;
                const float* vp = vbase + c * HW;        // L2-resident gathers
                const float  f  = vp[o00] * w00 + vp[o01] * w01
                                + vp[o10] * w10 + vp[o11] * w11;
                feats[n][i] = f;
                dot = fmaf(qv[i], f, dot);
            }
            // full-wave butterfly reduction: every lane gets the score
            #pragma unroll
            for (int off = 16; off > 0; off >>= 1)
                dot += __shfl_xor(dot, off, 32);
            score[n] = dot * 0.0625f;                    // 1/sqrt(256)
        }

        // softmax over 6 cameras (replicated in every lane)
        float mx = score[0];
        #pragma unroll
        for (int n = 1; n < N_CAM; ++n) mx = fmaxf(mx, score[n]);
        float e[N_CAM];
        float sum = 0.0f;
        #pragma unroll
        for (int n = 0; n < N_CAM; ++n) { e[n] = __expf(score[n] - mx); sum += e[n]; }
        const float inv = 1.0f / sum;

        // ctx = sum_n attn[n] * feats[n]  -> LDS (conflict-free writes)
        #pragma unroll
        for (int i = 0; i < 8; ++i) {
            float acc = 0.0f;
            #pragma unroll
            for (int n = 0; n < N_CAM; ++n) acc = fmaf(e[n] * inv, feats[n][i], acc);
            ctx[qlocal][i * 32 + lane] = acc;
        }
    }

    __syncthreads();

    // ---------------- Phase 2: output projection via WMMA ----------------
    // out[l][j] = sum_k ctx[l][k] * W_out[j][k] + b_out[j]
    const int m    = lane & 15;           // WMMA row (A-matrix M, per ISA layout)
    const int ncol = lane & 15;           // WMMA col (B-matrix N)
    const int koff = (lane >> 4) * 2;     // lanes 16-31 hold K=2,3 (K=10,11 etc. in f16 path)

    #pragma unroll
    for (int jj = 0; jj < 2; ++jj) {
        const int j0 = (wave + jj * 8) * 16;
        v8f acc = {};

#if __has_builtin(__builtin_amdgcn_wmma_f32_16x16x4_f32)
        // Full-precision fp32 WMMA: D(16x16) += A(16x4,f32) x B(4x16,f32)
        #pragma unroll 4
        for (int k = 0; k < CH; k += 4) {
            v2f a, b;
            a.x = ctx[m][k + koff];
            a.y = ctx[m][k + koff + 1];
            const float* wr = W_out + (j0 + ncol) * CH + k + koff;  // B = W_out^T
            b.x = wr[0];
            b.y = wr[1];
            acc = __builtin_amdgcn_wmma_f32_16x16x4_f32(
                      false, a, false, b, (short)0, acc, false, false);
        }
#else
        // Fallback: codegen-confirmed f16 WMMA (f32 accumulate)
        const int kb = (lane >> 4) * 8;   // 16-bit A/B layout: lanes 16-31 at K+8
        for (int k0 = 0; k0 < CH; k0 += 32) {
            v16h a, b;
            #pragma unroll
            for (int h = 0; h < 16; ++h) {
                const int k = k0 + ((h >> 3) * 16) + (h & 7) + kb;
                a[h] = (_Float16)ctx[m][k];
                b[h] = (_Float16)W_out[(j0 + ncol) * CH + k];
            }
            acc = __builtin_amdgcn_wmma_f32_16x16x32_f16(
                      false, a, false, b, (short)0, acc, false, false);
        }
#endif
        const float bias  = b_out[j0 + ncol];
        const int   rowhi = (lane >> 4) * 8;   // D layout: lanes 16-31 hold M=8..15
        #pragma unroll
        for (int r = 0; r < 8; ++r)
            out[(l0 + rowhi + r) * CH + j0 + ncol] = acc[r] + bias;
    }
}

extern "C" void kernel_launch(void* const* d_in, const int* in_sizes, int n_in,
                              void* d_out, int out_size, void* d_ws, size_t ws_size,
                              hipStream_t stream) {
    (void)in_sizes; (void)n_in; (void)out_size; (void)d_ws; (void)ws_size;
    const float* query    = (const float*)d_in[0];
    // d_in[1] = key (unused by reference)
    const float* value    = (const float*)d_in[2];
    // d_in[3] = query_pos (unused by reference)
    const float* refpts   = (const float*)d_in[4];
    const float* bev_mask = (const float*)d_in[5];
    const float* W_out    = (const float*)d_in[6];
    const float* b_out    = (const float*)d_in[7];
    float*       out      = (float*)d_out;

    dim3 grid(L_Q / QPB);   // 2500 blocks, 16 queries each
    sca_fused_kernel<<<grid, 256, 0, stream>>>(query, value, refpts, bev_mask,
                                               W_out, b_out, out);
}
